// MultiHeadAttention_15642270892248
// MI455X (gfx1250) — matrix operand order repaired and checked by reading, not compiled, not run on hardware
//
#include <hip/hip_runtime.h>
#include <hip/hip_bf16.h>

// ---------------------------------------------------------------------------
// MHA forward for MI455X (gfx1250, wave32, WMMA).
// qkv = x @ Wqkv^T + b ; flash-attention softmax(QK^T/sqrt(64))V ; out = ctx @ Wo^T + b
// All matmuls via v_wmma_f32_16x16x32_bf16 (fp32 accumulate).
// ---------------------------------------------------------------------------

typedef __bf16 bf16_t;
typedef __attribute__((ext_vector_type(16))) __bf16 v16bf;
typedef __attribute__((ext_vector_type(8)))  __bf16 v8bf;
typedef __attribute__((ext_vector_type(8)))  float  v8f;

static constexpr int Bn  = 2;
static constexpr int Ln  = 2048;
static constexpr int Dn  = 1024;
static constexpr int Hn  = 16;
static constexpr int DHn = 64;
static constexpr int M_ALL = Bn * Ln;     // 4096
static constexpr int N_QKV = 3 * Dn;      // 3072

// A-fragment (16x32 bf16, M x K): lane m = lane&15, hi = lane>>4.
// elems 0..15 <- row[hi*16 + 0..15]     (K = hi*16 + e)
__device__ inline v16bf load_a_frag(const bf16_t* row, int hi) {
    return *(const v16bf*)(row + hi * 16);
}

// B-fragment (32x16 bf16, K x N): lane n = lane&15, hi = lane>>4 holds the
// contiguous column slice col[hi*16 + 0..15].
__device__ inline v16bf load_b_frag(const bf16_t* col, int hi) {
    return *(const v16bf*)(col + hi * 16);
}

__device__ inline v8f wmma_bf16(v16bf a, v16bf b, v8f c) {
    return __builtin_amdgcn_wmma_f32_16x16x32_bf16(
        /*neg_a=*/false, a, /*neg_b=*/false, b,
        /*c_mod=*/(short)0, c, /*reuse_a=*/false, /*reuse_b=*/false);
}

// ---------------------------------------------------------------------------
__global__ void f32_to_bf16_kernel(const float* __restrict__ s,
                                   bf16_t* __restrict__ d, long n) {
    long i = (long)blockIdx.x * blockDim.x + threadIdx.x;
    long stride = (long)gridDim.x * blockDim.x;
    for (; i < n; i += stride) d[i] = (bf16_t)s[i];
}

// ---------------------------------------------------------------------------
// C = A(MxK) @ W(NxK)^T + bias.  Block: 256 thr = 8 waves, tile 64(M)x128(N);
// wave tile 16x64 (4 accumulators). K-loop step 32.
// mode 0: store fp32 to outF (ld = N)
// mode 1: qkv epilogue: n < 2D -> bf16 into outB (ld = N); n >= 2D -> bf16
//         transposed into vt[b][h][dh][l] (per-lane contiguous 16B store).
__global__ __launch_bounds__(256)
void gemm_bf16_wmma(const bf16_t* __restrict__ A, const bf16_t* __restrict__ W,
                    const float* __restrict__ bias, int M, int N, int K,
                    float* __restrict__ outF, bf16_t* __restrict__ outB,
                    bf16_t* __restrict__ vt, int mode) {
    const int tid  = threadIdx.x;
    const int wave = tid >> 5, lane = tid & 31;
    const int ln = lane & 15, hi = lane >> 4;
    const int wm = wave & 3, wn = wave >> 2;
    const int m0 = blockIdx.y * 64 + wm * 16;
    const int n0 = blockIdx.x * 128 + wn * 64;
    if (m0 >= M || n0 >= N) return;

    v8f acc[4];
#pragma unroll
    for (int nc = 0; nc < 4; ++nc)
#pragma unroll
        for (int r = 0; r < 8; ++r) acc[nc][r] = 0.0f;

    const bf16_t* arow = A + (size_t)(m0 + ln) * K;
    for (int kk = 0; kk < K; kk += 32) {
        v16bf a = load_a_frag(arow + kk, hi);
#pragma unroll
        for (int nc = 0; nc < 4; ++nc) {
            const bf16_t* wrow = W + (size_t)(n0 + nc * 16 + ln) * K + kk;
            acc[nc] = wmma_bf16(a, load_b_frag(wrow, hi), acc[nc]);
        }
    }

    // Epilogue. C layout: VGPR r -> row m0 + hi*8 + r, col n0 + nc*16 + ln.
#pragma unroll
    for (int nc = 0; nc < 4; ++nc) {
        const int n = n0 + nc * 16 + ln;
        const float bv = bias[n];
        if (mode == 0) {
#pragma unroll
            for (int r = 0; r < 8; ++r) {
                int m = m0 + hi * 8 + r;
                outF[(size_t)m * N + n] = acc[nc][r] + bv;
            }
        } else {
            if (n < 2 * Dn) {  // Q and K columns: row-major bf16
#pragma unroll
                for (int r = 0; r < 8; ++r) {
                    int m = m0 + hi * 8 + r;
                    outB[(size_t)m * N + n] = (bf16_t)(acc[nc][r] + bv);
                }
            } else {           // V columns: transposed store Vt[b][h][dh][l]
                int h  = (n - 2 * Dn) >> 6;
                int dh = (n - 2 * Dn) & 63;
                int b_ = m0 / Ln;               // 16-row tile never crosses b
                int l0 = (m0 & (Ln - 1)) + hi * 8;
                v8bf pk;
#pragma unroll
                for (int r = 0; r < 8; ++r) pk[r] = (bf16_t)(acc[nc][r] + bv);
                *(v8bf*)(vt + ((size_t)((b_ * Hn + h) * DHn + dh)) * Ln + l0) = pk;
            }
        }
    }
}

// ---------------------------------------------------------------------------
// Flash attention. Block: 128 thr = 4 waves; wave owns 16 queries of (b,h).
// Key blocks of 32: S(16x32) via 4 WMMAs, online softmax (shuffle-tree row
// max/sum within 16-lane groups), P -> LDS -> A-frag, PV via 4 WMMAs into acc.
__global__ __launch_bounds__(128)
void attn_fwd_wmma(const bf16_t* __restrict__ qkv, const bf16_t* __restrict__ vt,
                   const int* __restrict__ mask, bf16_t* __restrict__ ctx,
                   float scale) {
    __shared__ bf16_t plds[4][16][32];   // per-wave P staging tile

    const int tid  = threadIdx.x;
    const int wave = tid >> 5, lane = tid & 31;
    const int ln = lane & 15, hi = lane >> 4;
    const int q0 = blockIdx.x * 64 + wave * 16;
    const int bh = blockIdx.y;
    const int b  = bh >> 4, h = bh & 15;

    // Q fragments (16 x 64 = two K-steps of 32)
    const bf16_t* qrow = qkv + (size_t)(b * Ln + q0 + ln) * N_QKV + h * DHn;
    v16bf aq0 = load_a_frag(qrow, hi);
    v16bf aq1 = load_a_frag(qrow + 32, hi);

    v8f acc[4];
    float mi[8], li[8];
#pragma unroll
    for (int nc = 0; nc < 4; ++nc)
#pragma unroll
        for (int r = 0; r < 8; ++r) acc[nc][r] = 0.0f;
#pragma unroll
    for (int r = 0; r < 8; ++r) { mi[r] = -1e30f; li[r] = 0.0f; }

    for (int key0 = 0; key0 < Ln; key0 += 32) {
        // ---- S = Q @ K^T (16 x 32) : columns of K^T are contiguous K rows
        v8f s[2];
#pragma unroll
        for (int nc = 0; nc < 2; ++nc) {
            const bf16_t* kr = qkv + (size_t)(b * Ln + key0 + nc * 16 + ln) * N_QKV
                             + Dn + h * DHn;
            v8f sc;
#pragma unroll
            for (int r = 0; r < 8; ++r) sc[r] = 0.0f;
            sc = wmma_bf16(aq0, load_b_frag(kr, hi), sc);
            sc = wmma_bf16(aq1, load_b_frag(kr + 32, hi), sc);
            s[nc] = sc;
        }

        // ---- scale + mask
#pragma unroll
        for (int nc = 0; nc < 2; ++nc)
#pragma unroll
            for (int r = 0; r < 8; ++r) {
                float v = s[nc][r] * scale;
                int qi = q0 + hi * 8 + r;
                int ki = key0 + nc * 16 + ln;
                if (mask[(size_t)qi * Ln + ki] == 0) v = -1e30f;
                s[nc][r] = v;
            }

        // ---- online softmax (element r owns row hi*8+r; reduce over 16 lanes)
        float corr[8];
#pragma unroll
        for (int r = 0; r < 8; ++r) {
            float mx = fmaxf(s[0][r], s[1][r]);
            mx = fmaxf(mx, __shfl_xor(mx, 1, 32));
            mx = fmaxf(mx, __shfl_xor(mx, 2, 32));
            mx = fmaxf(mx, __shfl_xor(mx, 4, 32));
            mx = fmaxf(mx, __shfl_xor(mx, 8, 32));
            float mnew = fmaxf(mi[r], mx);
            corr[r] = __expf(mi[r] - mnew);
            float p0 = __expf(s[0][r] - mnew);
            float p1 = __expf(s[1][r] - mnew);
            s[0][r] = p0; s[1][r] = p1;
            float rs = p0 + p1;
            rs += __shfl_xor(rs, 1, 32);
            rs += __shfl_xor(rs, 2, 32);
            rs += __shfl_xor(rs, 4, 32);
            rs += __shfl_xor(rs, 8, 32);
            li[r] = li[r] * corr[r] + rs;
            mi[r] = mnew;
        }
#pragma unroll
        for (int nc = 0; nc < 4; ++nc)
#pragma unroll
            for (int r = 0; r < 8; ++r) acc[nc][r] *= corr[r];

        // ---- P (C layout) -> LDS -> A-fragment layout (in-wave, dscnt only)
#pragma unroll
        for (int nc = 0; nc < 2; ++nc)
#pragma unroll
            for (int r = 0; r < 8; ++r)
                plds[wave][hi * 8 + r][nc * 16 + ln] = (bf16_t)s[nc][r];
        asm volatile("s_wait_dscnt 0" ::: "memory");
        v16bf pa = load_a_frag(&plds[wave][ln][0], hi);

        // ---- acc += P @ V  (B-frag columns contiguous thanks to Vt layout)
        const bf16_t* vbase = vt + (size_t)((b * Hn + h) * DHn) * Ln + key0;
#pragma unroll
        for (int nc = 0; nc < 4; ++nc) {
            const bf16_t* vp = vbase + (size_t)(nc * 16 + ln) * Ln;
            acc[nc] = wmma_bf16(pa, load_b_frag(vp, hi), acc[nc]);
        }
    }

    // ---- normalize and store ctx (bf16, row-major B*L x D)
#pragma unroll
    for (int nc = 0; nc < 4; ++nc)
#pragma unroll
        for (int r = 0; r < 8; ++r) {
            float v = acc[nc][r] / li[r];
            ctx[(size_t)(b * Ln + q0 + hi * 8 + r) * Dn + h * DHn + nc * 16 + ln]
                = (bf16_t)v;
        }
}

// ---------------------------------------------------------------------------
extern "C" void kernel_launch(void* const* d_in, const int* in_sizes, int n_in,
                              void* d_out, int out_size, void* d_ws, size_t ws_size,
                              hipStream_t stream) {
    (void)in_sizes; (void)n_in; (void)out_size;
    const float* x    = (const float*)d_in[0];
    const int*   mask = (const int*)d_in[1];
    const float* Wqkv = (const float*)d_in[2];
    const float* bqkv = (const float*)d_in[3];
    const float* Wo   = (const float*)d_in[4];
    const float* bo   = (const float*)d_in[5];
    float* out = (float*)d_out;

    char* ws = (char*)d_ws;
    size_t off = 0;
    auto take = [&](size_t bytes) {
        char* p = ws + off;
        off += (bytes + 255) & ~(size_t)255;
        return p;
    };
    bf16_t* xbf    = (bf16_t*)take((size_t)M_ALL * Dn * 2);      // 8 MiB
    bf16_t* wqkvbf = (bf16_t*)take((size_t)N_QKV * Dn * 2);      // 6 MiB
    bf16_t* wobf   = (bf16_t*)take((size_t)Dn * Dn * 2);         // 2 MiB
    bf16_t* qkvbf  = (bf16_t*)take((size_t)M_ALL * N_QKV * 2);   // 24 MiB (Q,K)
    bf16_t* vtbf   = (bf16_t*)take((size_t)Bn * Hn * DHn * Ln * 2); // 8 MiB
    bf16_t* ctxbf  = (bf16_t*)take((size_t)M_ALL * Dn * 2);      // 8 MiB
    if (off > ws_size) return;  // insufficient workspace

    // 1) fp32 -> bf16 conversions
    f32_to_bf16_kernel<<<1024, 256, 0, stream>>>(x, xbf, (long)M_ALL * Dn);
    f32_to_bf16_kernel<<<1024, 256, 0, stream>>>(Wqkv, wqkvbf, (long)N_QKV * Dn);
    f32_to_bf16_kernel<<<512, 256, 0, stream>>>(Wo, wobf, (long)Dn * Dn);

    // 2) QKV projection (mode 1: bf16 Q/K + transposed V epilogue)
    gemm_bf16_wmma<<<dim3(N_QKV / 128, M_ALL / 64), 256, 0, stream>>>(
        xbf, wqkvbf, bqkv, M_ALL, N_QKV, Dn, nullptr, qkvbf, vtbf, 1);

    // 3) flash attention
    attn_fwd_wmma<<<dim3(Ln / 64, Bn * Hn), 128, 0, stream>>>(
        qkvbf, vtbf, mask, ctxbf, 0.125f /* 1/sqrt(64) */);

    // 4) output projection (mode 0: fp32 out)
    gemm_bf16_wmma<<<dim3(Dn / 128, M_ALL / 64), 256, 0, stream>>>(
        ctxbf, wobf, bo, M_ALL, Dn, Dn, out, nullptr, nullptr, 0);
}
